// HybridRegressionModel_65481071401506
// MI455X (gfx1250) — matrix operand order repaired and checked by reading, not compiled, not run on hardware
//
#include <hip/hip_runtime.h>
#include <hip/hip_bf16.h>

typedef __attribute__((ext_vector_type(16))) __bf16 v16bf;
typedef __attribute__((ext_vector_type(8)))  float  v8f;
typedef __attribute__((ext_vector_type(4)))  float  v4f;   // clang vector: OK for nontemporal builtins
typedef unsigned short u16;
typedef unsigned int   u32;

#define BM 128          // batch rows per workgroup
#define THREADS 256     // 8 waves (wave32)
#define NKC (2048 / 32) // 64 K-chunks

__device__ __forceinline__ u16 f2bf(float f) {
    u32 u = __builtin_bit_cast(u32, f);
    u += 0x7FFFu + ((u >> 16) & 1u);       // round-to-nearest-even
    return (u16)(u >> 16);
}

__device__ __forceinline__ u32 f2bf_pk(float a, float b) {
    return (u32)f2bf(a) | ((u32)f2bf(b) << 16);
}

__device__ __forceinline__ v16bf as_bf16x16(uint4 lo, uint4 hi) {
    struct P { uint4 a, b; } p{lo, hi};
    return __builtin_bit_cast(v16bf, p);
}

__global__ __launch_bounds__(THREADS, 2)
void hybrid_regression_kernel(const float* __restrict__ x,
                              const float* __restrict__ W1,
                              const float* __restrict__ b1,
                              const float* __restrict__ W2,
                              const float* __restrict__ b2,
                              const float* __restrict__ W3,
                              const float* __restrict__ b3,
                              const float* __restrict__ conv_k,
                              const float* __restrict__ conv_b,
                              const float* __restrict__ basis,
                              const float* __restrict__ head_w,
                              const float* __restrict__ head_b,
                              float* __restrict__ out)
{
    // ---------------- LDS ----------------
    __shared__ u16   sA [128 * 32];     // x chunk, bf16 row-major   [m][k]   8 KB
    __shared__ u16   sW1[128 * 32];     // W1 chunk, bf16 col-major  [n][k]   8 KB
    __shared__ u16   sA2[128 * 128];    // h1 bf16 row-major         [m][k]  32 KB
    __shared__ u16   sW2[ 64 * 128];    // W2 bf16 col-major         [n][k]  16 KB
    __shared__ float sH2[128 * 64];     // h2 fp32                           32 KB
    __shared__ float sW3[64 * 4];       // W3 fp32                            1 KB
    __shared__ float sZ [128 * 6];      // z0..z3, conv, |z|^2                3 KB
    __shared__ float sBasis[1024 * 5];  //                                   20 KB
    __shared__ float sBnorm[1024];      // |basis_j|^2                        4 KB
    __shared__ float sHw[1024];         // head_w                             4 KB
    __shared__ float sPartial[256];     //                                    1 KB

    const int tid  = threadIdx.x;
    const int wave = tid >> 5;          // 0..7
    const int lane = tid & 31;
    const int lm   = lane & 15;
    const int hi   = lane >> 4;         // half-wave selector
    const int rowBase = blockIdx.x * BM;

    // ---------------- stage constants (consumed after later barriers) ----
    // W2 [128][64] f32 -> sW2 [n][k] bf16  (8192 floats = 2048 float4)
    #pragma unroll
    for (int i = 0; i < 8; ++i) {
        int idx = tid + THREADS * i;          // 0..2047
        int k   = idx >> 4;                   // 16 float4 per row of 64
        int c4  = idx & 15;
        v4f v = reinterpret_cast<const v4f*>(W2)[idx];
        int n = c4 * 4;
        sW2[(n + 0) * 128 + k] = f2bf(v.x);
        sW2[(n + 1) * 128 + k] = f2bf(v.y);
        sW2[(n + 2) * 128 + k] = f2bf(v.z);
        sW2[(n + 3) * 128 + k] = f2bf(v.w);
    }
    sW3[tid] = W3[tid];                       // 256 floats exactly
    #pragma unroll
    for (int r = 0; r < 4; ++r) {
        int j = tid * 4 + r;                  // 0..1023
        float s = 0.f;
        #pragma unroll
        for (int c = 0; c < 5; ++c) {
            float v = basis[j * 5 + c];
            sBasis[j * 5 + c] = v;
            s += v * v;
        }
        sBnorm[j] = s;
        sHw[j]    = head_w[j];
    }

    // ---------------- GEMM1: h1 = relu(x @ W1 + b1), K = 2048 ------------
    v8f acc1[8];
    #pragma unroll
    for (int nt = 0; nt < 8; ++nt) acc1[nt] = v8f{0.f,0.f,0.f,0.f,0.f,0.f,0.f,0.f};

    const int m = wave * 16 + lm;             // this wave's WMMA row (per lane)

    // per-thread coalesced load coordinates (fixed across chunks)
    int xr[4], xc[4], wk[4], wn[4];
    #pragma unroll
    for (int i = 0; i < 4; ++i) {
        int idx = tid + THREADS * i;          // 0..1023
        xr[i] = idx >> 3;                     // x: 8 float4 per 32-f32 row
        xc[i] = idx & 7;
        wk[i] = idx >> 5;                     // W1: 32 float4 per 128-f32 row
        wn[i] = (idx & 31) * 4;
    }

    // software pipeline: register-buffer next chunk while WMMAs run
    v4f xv[4], wv[4];
    #pragma unroll
    for (int i = 0; i < 4; ++i) {
        xv[i] = __builtin_nontemporal_load(reinterpret_cast<const v4f*>(
                    x + (size_t)(rowBase + xr[i]) * 2048) + xc[i]);   // kc = 0
        wv[i] = reinterpret_cast<const v4f*>(
                    W1 + (size_t)wk[i] * 128)[wn[i] >> 2];
    }

    for (int kc = 0; kc < NKC; ++kc) {
        __syncthreads();                      // prev-iter fragment reads done

        // commit register-staged chunk kc to LDS (bf16); waits only on loadcnt
        #pragma unroll
        for (int i = 0; i < 4; ++i) {
            int o = xr[i] * 32 + xc[i] * 4;
            *reinterpret_cast<u32*>(&sA[o + 0]) = f2bf_pk(xv[i].x, xv[i].y);
            *reinterpret_cast<u32*>(&sA[o + 2]) = f2bf_pk(xv[i].z, xv[i].w);
            int n = wn[i], k = wk[i];
            sW1[(n + 0) * 32 + k] = f2bf(wv[i].x);
            sW1[(n + 1) * 32 + k] = f2bf(wv[i].y);
            sW1[(n + 2) * 32 + k] = f2bf(wv[i].z);
            sW1[(n + 3) * 32 + k] = f2bf(wv[i].w);
        }
        __syncthreads();

        // issue global loads for chunk kc+1 AFTER the barrier: they stay
        // outstanding across the whole WMMA phase (consumed at next commit)
        if (kc + 1 < NKC) {
            #pragma unroll
            for (int i = 0; i < 4; ++i) {
                xv[i] = __builtin_nontemporal_load(reinterpret_cast<const v4f*>(
                            x + (size_t)(rowBase + xr[i]) * 2048 + (kc + 1) * 32) + xc[i]);
                wv[i] = reinterpret_cast<const v4f*>(
                            W1 + (size_t)((kc + 1) * 32 + wk[i]) * 128)[wn[i] >> 2];
            }
        }

        // A fragment: 16-bit A 16x32 layout -> two ds_load_b128 per lane
        v16bf a = as_bf16x16(*reinterpret_cast<uint4*>(&sA[m * 32 + hi * 8]),
                             *reinterpret_cast<uint4*>(&sA[m * 32 + 16 + hi * 8]));
        #pragma unroll
        for (int nt = 0; nt < 8; ++nt) {
            int n = nt * 16 + lm;             // 16-bit B 32x16 layout
            v16bf b = as_bf16x16(*reinterpret_cast<uint4*>(&sW1[n * 32 + hi * 8]),
                                 *reinterpret_cast<uint4*>(&sW1[n * 32 + 16 + hi * 8]));
            acc1[nt] = __builtin_amdgcn_wmma_f32_16x16x32_bf16(
                false, a, false, b, (short)0, acc1[nt], false, false);
        }
    }

    __syncthreads();
    // bias + relu, write h1 as bf16 A-layout for GEMM2
    #pragma unroll
    for (int nt = 0; nt < 8; ++nt) {
        int n = nt * 16 + lm;
        float bias = b1[n];
        #pragma unroll
        for (int r = 0; r < 8; ++r) {
            int row = wave * 16 + hi * 8 + r;   // C-layout: M = r + 8*hi
            float v = acc1[nt][r] + bias;
            v = v > 0.f ? v : 0.f;
            sA2[row * 128 + n] = f2bf(v);
        }
    }
    __syncthreads();

    // ---------------- GEMM2: h2 = relu(h1 @ W2 + b2), 128x64x128 ---------
    v8f acc2[4];
    #pragma unroll
    for (int nt = 0; nt < 4; ++nt) acc2[nt] = v8f{0.f,0.f,0.f,0.f,0.f,0.f,0.f,0.f};

    #pragma unroll
    for (int kc = 0; kc < 4; ++kc) {
        v16bf a = as_bf16x16(
            *reinterpret_cast<uint4*>(&sA2[m * 128 + kc * 32 + hi * 8]),
            *reinterpret_cast<uint4*>(&sA2[m * 128 + kc * 32 + 16 + hi * 8]));
        #pragma unroll
        for (int nt = 0; nt < 4; ++nt) {
            int n = nt * 16 + lm;
            v16bf b = as_bf16x16(
                *reinterpret_cast<uint4*>(&sW2[n * 128 + kc * 32 + hi * 8]),
                *reinterpret_cast<uint4*>(&sW2[n * 128 + kc * 32 + 16 + hi * 8]));
            acc2[nt] = __builtin_amdgcn_wmma_f32_16x16x32_bf16(
                false, a, false, b, (short)0, acc2[nt], false, false);
        }
    }
    #pragma unroll
    for (int nt = 0; nt < 4; ++nt) {
        int n = nt * 16 + lm;
        float bias = b2[n];
        #pragma unroll
        for (int r = 0; r < 8; ++r) {
            int row = wave * 16 + hi * 8 + r;
            float v = acc2[nt][r] + bias;
            sH2[row * 64 + n] = v > 0.f ? v : 0.f;
        }
    }
    __syncthreads();

    // ---------------- z = h2 @ W3 + b3 ; conv sigmoid ---------------------
    if (tid < 128) {
        float z0 = b3[0], z1 = b3[1], z2 = b3[2], z3 = b3[3];
        const float* hrow = &sH2[tid * 64];
        #pragma unroll 8
        for (int k = 0; k < 64; ++k) {
            float h = hrow[k];
            z0 += h * sW3[k * 4 + 0];
            z1 += h * sW3[k * 4 + 1];
            z2 += h * sW3[k * 4 + 2];
            z3 += h * sW3[k * 4 + 3];
        }
        float cv = z0 * conv_k[0] + z1 * conv_k[1] + z2 * conv_k[2] + z3 * conv_k[3]
                 + conv_b[0];                                // THRESHOLD == 0
        float sg = 1.f / (1.f + __expf(-cv));
        float zn = z0*z0 + z1*z1 + z2*z2 + z3*z3 + sg*sg;
        float* zr = &sZ[tid * 6];
        zr[0] = z0; zr[1] = z1; zr[2] = z2; zr[3] = z3; zr[4] = sg; zr[5] = zn;
    }
    __syncthreads();

    // ---------------- RBF + head: out = exp(-||z-b||^2) @ head_w + head_b -
    {
        int row  = tid >> 1;
        int half = tid & 1;
        const float* zr = &sZ[row * 6];
        float z0 = zr[0], z1 = zr[1], z2 = zr[2], z3 = zr[3], z4 = zr[4], zn = zr[5];
        float acc = 0.f;
        int j0 = half * 512;
        for (int j = j0; j < j0 + 512; ++j) {
            const float* bb = &sBasis[j * 5];
            float dot = z0*bb[0] + z1*bb[1] + z2*bb[2] + z3*bb[3] + z4*bb[4];
            float d   = zn + sBnorm[j] - 2.f * dot;          // GAMMA == 1
            acc += sHw[j] * __expf(-d);
        }
        sPartial[tid] = acc;
    }
    __syncthreads();
    if (tid < 128) {
        out[rowBase + tid] = sPartial[tid * 2] + sPartial[tid * 2 + 1] + head_b[0];
    }
}

extern "C" void kernel_launch(void* const* d_in, const int* in_sizes, int n_in,
                              void* d_out, int out_size, void* d_ws, size_t ws_size,
                              hipStream_t stream) {
    const float* x      = (const float*)d_in[0];
    const float* W1     = (const float*)d_in[1];
    const float* b1     = (const float*)d_in[2];
    const float* W2     = (const float*)d_in[3];
    const float* b2     = (const float*)d_in[4];
    const float* W3     = (const float*)d_in[5];
    const float* b3     = (const float*)d_in[6];
    const float* conv_k = (const float*)d_in[7];
    const float* conv_b = (const float*)d_in[8];
    const float* basis  = (const float*)d_in[9];
    const float* head_w = (const float*)d_in[10];
    const float* head_b = (const float*)d_in[11];
    float* out = (float*)d_out;

    const int batch = in_sizes[0] / 2048;     // 65536
    dim3 grid(batch / BM), block(THREADS);
    hipLaunchKernelGGL(hybrid_regression_kernel, grid, block, 0, stream,
                       x, W1, b1, W2, b2, W3, b3, conv_k, conv_b,
                       basis, head_w, head_b, out);
}
